// LSTM_LSTM_att_74603581932063
// MI455X (gfx1250) — matrix operand order repaired and checked by reading, not compiled, not run on hardware
//
#include <hip/hip_runtime.h>
#include <hip/hip_bf16.h>
#include <math.h>

// Problem constants (from reference)
#define Bsz 1024
#define Tsz 168
#define Esz 8
#define Dsz 4
#define Hsz 512
#define Osz 24
#define NG  2048           // 4*H gate width
#define KEXT 576           // H (512) + padded input chunk (64) -> 18 K-chunks (even)

typedef __attribute__((ext_vector_type(16))) __bf16 v16bf;
typedef __attribute__((ext_vector_type(8)))  __bf16 v8bf;
typedef __attribute__((ext_vector_type(8)))  float  v8f;

// ---------------------------------------------------------------------------
// Weight prep: Wext[n, 0:512] = Whh[n,:] ; Wext[n, 512:512+Ein] = Wih[n,:] ; rest 0
// ---------------------------------------------------------------------------
__global__ void prep_w_kernel(const float* __restrict__ Whh,
                              const float* __restrict__ Wih,
                              int Ein, __bf16* __restrict__ Wext) {
    int tid = blockIdx.x * blockDim.x + threadIdx.x;
    if (tid >= NG * KEXT) return;
    int n = tid / KEXT;
    int k = tid % KEXT;
    float v = 0.0f;
    if (k < Hsz)             v = Whh[n * Hsz + k];
    else if (k - Hsz < Ein)  v = Wih[n * Ein + (k - Hsz)];
    Wext[n * KEXT + k] = (__bf16)v;
}

// ---------------------------------------------------------------------------
// Init: h = c = 0, A head = 0, A tail = x[:, 0, :] zero-padded
// ---------------------------------------------------------------------------
__global__ void init_state_kernel(float* __restrict__ c, float* __restrict__ h,
                                  __bf16* __restrict__ A, const float* __restrict__ x) {
    int tid = blockIdx.x * blockDim.x + threadIdx.x;
    if (tid >= Bsz * KEXT) return;
    int m = tid / KEXT;
    int j = tid % KEXT;
    if (j < Hsz) {
        c[m * Hsz + j] = 0.0f;
        h[m * Hsz + j] = 0.0f;
        A[m * KEXT + j] = (__bf16)0.0f;
    } else {
        int k = j - Hsz;
        float v = (k < Esz) ? x[(m * Tsz + 0) * Esz + k] : 0.0f;
        A[m * KEXT + j] = (__bf16)v;
    }
}

// ---------------------------------------------------------------------------
// Core bf16 WMMA GEMM: gates(1024 x 2048, fp32) = A(1024 x 576, bf16)
//                      x Wext^T(576 x 2048, row-major N x K bf16)
// Block = 256 threads (8 waves), block tile 128x128, wave tile 32x64.
// Software-pipelined K loop: double-buffered fragments so next chunk's
// global loads are in flight while current chunk's 8 WMMAs issue.
// Fragment loads use a K-contiguous order, same for A and B (not AMD's table order):
//   A 16x32: lanes 0-15 -> M=lane, K0..15;                  lanes 16-31 -> K16..31
//   B 32x16: lanes 0-15 -> N=lane, K0..15;                  lanes 16-31 -> K16..31
// ---------------------------------------------------------------------------
__device__ __forceinline__ void load_frags(const __bf16* const* rowA,
                                           const __bf16* const* rowB,
                                           int k0, v16bf a[2], v16bf b[4]) {
#pragma unroll
    for (int i = 0; i < 2; ++i) {
        a[i] = *(const v16bf*)(rowA[i] + k0);
    }
#pragma unroll
    for (int j = 0; j < 4; ++j)
        b[j] = *(const v16bf*)(rowB[j] + k0);   // contiguous 32B chunk per lane
}

__global__ __launch_bounds__(256) void gemm_gates_kernel(const __bf16* __restrict__ A,
                                                         const __bf16* __restrict__ Bw,
                                                         float* __restrict__ gates) {
    const int wave  = threadIdx.x >> 5;
    const int lane  = threadIdx.x & 31;
    const int waveM = wave >> 1;                 // 0..3
    const int waveN = wave & 1;                  // 0..1
    const int m0 = blockIdx.x * 128 + waveM * 32;   // wave rows
    const int n0 = blockIdx.y * 128 + waveN * 64;   // wave cols
    const int laneHi = lane >> 4;                // 0/1
    const int laneLo = lane & 15;

    // Per-lane base row pointers (K offset applied in the loop)
    const __bf16* rowA[2];
    const __bf16* rowB[4];
#pragma unroll
    for (int i = 0; i < 2; ++i)
        rowA[i] = A + (size_t)(m0 + i * 16 + laneLo) * KEXT + laneHi * 16;
#pragma unroll
    for (int j = 0; j < 4; ++j)
        rowB[j] = Bw + (size_t)(n0 + j * 16 + laneLo) * KEXT + laneHi * 16;

    v8f acc[2][4];
#pragma unroll
    for (int i = 0; i < 2; ++i)
#pragma unroll
        for (int j = 0; j < 4; ++j)
            acc[i][j] = (v8f){0.f, 0.f, 0.f, 0.f, 0.f, 0.f, 0.f, 0.f};

    v16bf aP[2], bP[4];   // ping
    v16bf aQ[2], bQ[4];   // pong
    load_frags(rowA, rowB, 0, aP, bP);

#pragma unroll 1
    for (int k0 = 0; k0 < KEXT; k0 += 64) {
        // prefetch chunk k0+32 while computing on chunk k0
        load_frags(rowA, rowB, k0 + 32, aQ, bQ);
#pragma unroll
        for (int i = 0; i < 2; ++i)
#pragma unroll
            for (int j = 0; j < 4; ++j)
                acc[i][j] = __builtin_amdgcn_wmma_f32_16x16x32_bf16(
                    false, aP[i], false, bP[j], (short)0, acc[i][j], false, false);
        // prefetch chunk k0+64 (if any) while computing on chunk k0+32
        if (k0 + 64 < KEXT)
            load_frags(rowA, rowB, k0 + 64, aP, bP);
#pragma unroll
        for (int i = 0; i < 2; ++i)
#pragma unroll
            for (int j = 0; j < 4; ++j)
                acc[i][j] = __builtin_amdgcn_wmma_f32_16x16x32_bf16(
                    false, aQ[i], false, bQ[j], (short)0, acc[i][j], false, false);
    }

    // --- store: VGPR r, lanes 0-15 -> row r, lanes 16-31 -> row r+8 ---
#pragma unroll
    for (int i = 0; i < 2; ++i) {
        int rowBase = m0 + i * 16 + 8 * laneHi;
#pragma unroll
        for (int j = 0; j < 4; ++j) {
            int col = n0 + j * 16 + laneLo;
#pragma unroll
            for (int r = 0; r < 8; ++r)
                gates[(size_t)(rowBase + r) * NG + col] = acc[i][j][r];
        }
    }
}

// ---------------------------------------------------------------------------
// Fused LSTM cell (+ bias) ; encoder variant also refreshes A's input tail
// ---------------------------------------------------------------------------
__device__ __forceinline__ float sigmoidf_(float v) { return 1.0f / (1.0f + __expf(-v)); }

__device__ __forceinline__ void cell_body(int m, int j,
                                          const float* __restrict__ gates,
                                          const float* __restrict__ bias,
                                          float* __restrict__ c, float* __restrict__ h,
                                          __bf16* __restrict__ A) {
    const float* g = gates + m * NG;
    float gi = g[j]            + bias[j];
    float gf = g[j + Hsz]      + bias[j + Hsz];
    float gg = g[j + 2 * Hsz]  + bias[j + 2 * Hsz];
    float go = g[j + 3 * Hsz]  + bias[j + 3 * Hsz];
    float cc = c[m * Hsz + j];
    cc = sigmoidf_(gf) * cc + sigmoidf_(gi) * tanhf(gg);
    float hh = sigmoidf_(go) * tanhf(cc);
    c[m * Hsz + j] = cc;
    h[m * Hsz + j] = hh;
    A[m * KEXT + j] = (__bf16)hh;          // bf16 operand for next step's WMMA
}

__global__ void enc_cell_kernel(const float* __restrict__ gates,
                                const float* __restrict__ bias,
                                float* __restrict__ c, float* __restrict__ h,
                                __bf16* __restrict__ A,
                                const float* __restrict__ x, int t) {
    int tid = blockIdx.x * blockDim.x + threadIdx.x;
    if (tid >= Bsz * KEXT) return;
    int m = tid / KEXT;
    int j = tid % KEXT;
    if (j < Hsz) {
        cell_body(m, j, gates, bias, c, h, A);
    } else {
        int k = j - Hsz;
        float v = 0.0f;
        if (t + 1 < Tsz) {               // next encoder input x[:, t+1, :]
            if (k < Esz) v = x[(m * Tsz + t + 1) * Esz + k];
        } else {                          // seed decoder: xt0 = x[:, -1, -D:]
            if (k < Dsz) v = x[(m * Tsz + Tsz - 1) * Esz + (Esz - Dsz) + k];
        }
        A[m * KEXT + j] = (__bf16)v;
    }
}

__global__ void dec_cell_kernel(const float* __restrict__ gates,
                                const float* __restrict__ bias,
                                float* __restrict__ c, float* __restrict__ h,
                                __bf16* __restrict__ A) {
    int tid = blockIdx.x * blockDim.x + threadIdx.x;
    if (tid >= Bsz * Hsz) return;
    cell_body(tid / Hsz, tid % Hsz, gates, bias, c, h, A);
}

// ---------------------------------------------------------------------------
// Decoder FC: y = h @ W_fc^T + b_fc ; feed y back into A tail, emit y[:, -1]
// ---------------------------------------------------------------------------
__global__ void dec_fc_kernel(const float* __restrict__ h,
                              const float* __restrict__ Wfc,
                              const float* __restrict__ bfc,
                              __bf16* __restrict__ A,
                              float* __restrict__ out, int step) {
    int tid = blockIdx.x * blockDim.x + threadIdx.x;
    if (tid >= Bsz * Dsz) return;
    int m = tid / Dsz;
    int d = tid % Dsz;
    float s = bfc[d];
    const float* hr = h + m * Hsz;
    const float* wr = Wfc + d * Hsz;
#pragma unroll 8
    for (int k = 0; k < Hsz; ++k) s += hr[k] * wr[k];
    A[m * KEXT + Hsz + d] = (__bf16)s;               // next decoder input
    if (d == Dsz - 1) out[m * Osz + step] = s;       // ys = y[:, -1]
}

// ---------------------------------------------------------------------------
// Host orchestration (graph-capturable: everything on `stream`)
// ---------------------------------------------------------------------------
extern "C" void kernel_launch(void* const* d_in, const int* in_sizes, int n_in,
                              void* d_out, int out_size, void* d_ws, size_t ws_size,
                              hipStream_t stream) {
    const float* x       = (const float*)d_in[0];   // (B, T, E)
    const float* Wih_enc = (const float*)d_in[1];   // (4H, E)
    const float* Whh_enc = (const float*)d_in[2];   // (4H, H)
    const float* b_enc   = (const float*)d_in[3];   // (4H,)
    const float* Wih_dec = (const float*)d_in[4];   // (4H, D)
    const float* Whh_dec = (const float*)d_in[5];   // (4H, H)
    const float* b_dec   = (const float*)d_in[6];   // (4H,)
    const float* W_fc    = (const float*)d_in[7];   // (D, H)
    const float* b_fc    = (const float*)d_in[8];   // (D,)
    float* out = (float*)d_out;                     // (B, O, 1) fp32

    // Workspace carve (256B aligned slabs)
    char* w = (char*)d_ws;
    size_t off = 0;
    auto carve = [&](size_t bytes) {
        void* p = w + off;
        off += (bytes + 255) & ~size_t(255);
        return p;
    };
    __bf16* A     = (__bf16*)carve((size_t)Bsz * KEXT * sizeof(__bf16)); // [h | x_t] bf16
    float*  gates = (float*) carve((size_t)Bsz * NG * sizeof(float));
    float*  hbuf  = (float*) carve((size_t)Bsz * Hsz * sizeof(float));
    float*  cbuf  = (float*) carve((size_t)Bsz * Hsz * sizeof(float));
    __bf16* Wenc  = (__bf16*)carve((size_t)NG * KEXT * sizeof(__bf16));  // [Whh|Wih] bf16
    __bf16* Wdec  = (__bf16*)carve((size_t)NG * KEXT * sizeof(__bf16));

    const int TPB = 256;
    // One-time (per call) weight conversion to L2-resident bf16 images
    prep_w_kernel<<<(NG * KEXT + TPB - 1) / TPB, TPB, 0, stream>>>(Whh_enc, Wih_enc, Esz, Wenc);
    prep_w_kernel<<<(NG * KEXT + TPB - 1) / TPB, TPB, 0, stream>>>(Whh_dec, Wih_dec, Dsz, Wdec);
    init_state_kernel<<<(Bsz * KEXT + TPB - 1) / TPB, TPB, 0, stream>>>(cbuf, hbuf, A, x);

    dim3 ggrid(Bsz / 128, NG / 128);   // 8 x 16 workgroups

    // Encoder: 168 sequential steps
    for (int t = 0; t < Tsz; ++t) {
        gemm_gates_kernel<<<ggrid, TPB, 0, stream>>>(A, Wenc, gates);
        enc_cell_kernel<<<(Bsz * KEXT + TPB - 1) / TPB, TPB, 0, stream>>>(
            gates, b_enc, cbuf, hbuf, A, x, t);
    }
    // Decoder: 24 sequential steps
    for (int s = 0; s < Osz; ++s) {
        gemm_gates_kernel<<<ggrid, TPB, 0, stream>>>(A, Wdec, gates);
        dec_cell_kernel<<<(Bsz * Hsz + TPB - 1) / TPB, TPB, 0, stream>>>(
            gates, b_dec, cbuf, hbuf, A);
        dec_fc_kernel<<<(Bsz * Dsz + TPB - 1) / TPB, TPB, 0, stream>>>(
            hbuf, W_fc, b_fc, A, out, s);
    }
}
